// QQwen3MLP_79912161509833
// MI455X (gfx1250) — compile-verified, matrix operand order checked by reading, not run on hardware
//
#include <hip/hip_runtime.h>

// ---------------------------------------------------------------------------
// Quantized Qwen3 MLP for MI455X (gfx1250), wave32.
//  - GEMMs on v_wmma_f32_16x16x32_f16
//  - Global->LDS tile movement via GLOBAL_LOAD_ASYNC_TO_LDS_B128 (ASYNCcnt),
//    double-buffered so DMA of tile k+1 overlaps WMMA on tile k.
//  - perm cancels inside the GEMMs; reduced to an outlier byte-mask.
// ---------------------------------------------------------------------------

typedef __attribute__((ext_vector_type(16))) _Float16 v16h;
typedef __attribute__((ext_vector_type(8)))  float    v8f;

#define TPB 256
#define BM 128        // block M tile
#define BN 64         // block N tile
#define BK 32         // K step (one WMMA K)
#define LDS_STRIDE 40 // padded row stride in halves (80B) -> bank-spread frags
#define GROUP_M 16    // L2-aware block swizzle group

// ---------------- gfx1250 async Global->LDS helpers ----------------

__device__ __forceinline__ unsigned int lds_off(const void* p) {
  // LDS aperture: flat addr[31:0] == wave-relative LDS byte address
  return (unsigned int)(unsigned long long)p;
}

__device__ __forceinline__ void async_ld_b128(unsigned int lds_byte,
                                              const _Float16* gaddr) {
  asm volatile("global_load_async_to_lds_b128 %0, %1, off"
               :: "v"(lds_byte), "v"(gaddr) : "memory");
}

#define WAIT_ASYNC(imm) asm volatile("s_wait_asynccnt " #imm ::: "memory")

// ---------------- WMMA fragment helpers ----------------

__device__ __forceinline__ v16h ld_frag(const _Float16* p) {
  // 16-bit A/B 16x32 layout: lane holds halves [kb..kb+7] and [kb+16..kb+23]
  union { v16h v; uint4 q[2]; } u;
  u.q[0] = *(const uint4*)(p);
  u.q[1] = *(const uint4*)(p + 16);
  return u.v;
}

__device__ __forceinline__ void swizzle_pid(int pid, int nPidM, int nPidN,
                                            int& pm, int& pn) {
  int ppg   = GROUP_M * nPidN;
  int gid   = pid / ppg;
  int first = gid * GROUP_M;
  int gsz   = nPidM - first; if (gsz > GROUP_M) gsz = GROUP_M;
  int pin   = pid % ppg;
  pm = first + (pin % gsz);
  pn = pin / gsz;
}

// ---------------- prologue kernels ----------------

__global__ void f32_to_f16_kernel(const float* __restrict__ s,
                                  _Float16* __restrict__ d, size_t n) {
  size_t i = (size_t)blockIdx.x * blockDim.x + threadIdx.x;
  size_t stride = (size_t)gridDim.x * blockDim.x;
  for (; i < n; i += stride) d[i] = (_Float16)s[i];
}

__global__ void build_masks_kernel(const int* __restrict__ perm_up,
                                   const int* __restrict__ perm_down,
                                   const int* __restrict__ sel_up_p,
                                   const int* __restrict__ sel_down_p,
                                   unsigned char* __restrict__ mu,
                                   unsigned char* __restrict__ md,
                                   int D, int F) {
  int su = sel_up_p[0], sd = sel_down_p[0];
  for (int i = threadIdx.x; i < su; i += blockDim.x) mu[perm_up[D - su + i]] = 1;
  for (int i = threadIdx.x; i < sd; i += blockDim.x) md[perm_down[F - sd + i]] = 1;
}

// per-token masked absmax + int8 quant-dequant of x -> fp16
__global__ __launch_bounds__(TPB)
void quant_x_kernel(const float* __restrict__ x,
                    const unsigned char* __restrict__ mask,
                    _Float16* __restrict__ out, int D) {
  __shared__ float sx[4096];
  __shared__ float red[TPB];
  int n = blockIdx.x, tid = threadIdx.x;
  const float* xr = x + (size_t)n * D;
  float amax = 0.f;
  for (int i = tid; i < D; i += TPB) {
    float v = xr[i];
    sx[i] = v;
    if (!mask[i]) amax = fmaxf(amax, fabsf(v));
  }
  red[tid] = amax; __syncthreads();
  for (int s = TPB / 2; s > 0; s >>= 1) {
    if (tid < s) red[tid] = fmaxf(red[tid], red[tid + s]);
    __syncthreads();
  }
  float scale = fmaxf(red[0], 1e-8f) * (1.0f / 127.0f);
  float inv   = 1.0f / scale;
  for (int i = tid; i < D; i += TPB) {
    float v = sx[i];
    float q = fminf(fmaxf(rintf(v * inv), -127.f), 127.f) * scale;
    out[(size_t)n * D + i] = (_Float16)(mask[i] ? v : q);
  }
}

// per-token masked absmax + quant-dequant of h (fp16, in place)
__global__ __launch_bounds__(TPB)
void quant_h_kernel(_Float16* __restrict__ h,
                    const unsigned char* __restrict__ mask, int F) {
  __shared__ _Float16 sh[12288];
  __shared__ float red[TPB];
  int n = blockIdx.x, tid = threadIdx.x;
  _Float16* hr = h + (size_t)n * F;
  float amax = 0.f;
  for (int i = tid; i < F; i += TPB) {
    _Float16 hv = hr[i];
    sh[i] = hv;
    if (!mask[i]) amax = fmaxf(amax, fabsf((float)hv));
  }
  red[tid] = amax; __syncthreads();
  for (int s = TPB / 2; s > 0; s >>= 1) {
    if (tid < s) red[tid] = fmaxf(red[tid], red[tid + s]);
    __syncthreads();
  }
  float scale = fmaxf(red[0], 1e-8f) * (1.0f / 127.0f);
  float inv   = 1.0f / scale;
  for (int i = tid; i < F; i += TPB) {
    float v = (float)sh[i];
    float q = fminf(fmaxf(rintf(v * inv), -127.f), 127.f) * scale;
    hr[i] = (_Float16)(mask[i] ? v : q);
  }
}

// ---------------- GEMM 1: h = silu(xq@Wg^T) * (xq@Wu^T), fp16 out ----------
// Per issue, every wave posts exactly 4 async b128 (2 for A, 2 for B tile).

__global__ __launch_bounds__(TPB)
void gemm1_dual_silu(const _Float16* __restrict__ A,   // [M,K] xq
                     const _Float16* __restrict__ Bg,  // [N,K] w_gate fp16
                     const _Float16* __restrict__ Bu,  // [N,K] w_up fp16
                     _Float16* __restrict__ H,         // [M,N]
                     int M, int Ndim, int K) {
  __shared__ _Float16 sA [2][BM * LDS_STRIDE];
  __shared__ _Float16 sBg[2][BN * LDS_STRIDE];
  __shared__ _Float16 sBu[2][BN * LDS_STRIDE];

  int pm, pn;
  swizzle_pid(blockIdx.x, M / BM, Ndim / BN, pm, pn);
  int m0 = pm * BM, n0 = pn * BN;

  int tid = threadIdx.x, lane = tid & 31, wid = tid >> 5;
  int wm = (wid >> 1) * 32;     // 4 wave-rows of 32
  int wn = (wid & 1) * 32;      // 2 wave-cols of 32

  const v8f zero8 = {0.f,0.f,0.f,0.f,0.f,0.f,0.f,0.f};
  v8f accG[2][2], accU[2][2];
  for (int i = 0; i < 2; ++i)
    for (int j = 0; j < 2; ++j) { accG[i][j] = zero8; accU[i][j] = zero8; }

  // loaders: thread t owns row t>>1, halves [(t&1)*16 .. +15]  (32B = 2xb128)
  int ldRow = tid >> 1;                 // 0..127
  int ldOff = (tid & 1) * 16;           // halves
  const _Float16* gA = A + (size_t)(m0 + ldRow) * K + ldOff;
  int bRow = (tid < 128) ? ldRow : (ldRow - 64);   // wave-uniform branch
  const _Float16* gB = (tid < 128)
      ? (Bg + (size_t)(n0 + bRow) * K + ldOff)
      : (Bu + (size_t)(n0 + bRow) * K + ldOff);

  int fr = lane & 15;                   // frag row/col
  int kb = (lane >> 4) * 8;             // frag K base (halves)

#define G1_ISSUE(buf, k)                                                     \
  do {                                                                       \
    async_ld_b128(lds_off(&sA[buf][ldRow * LDS_STRIDE + ldOff]),     gA + (k));      \
    async_ld_b128(lds_off(&sA[buf][ldRow * LDS_STRIDE + ldOff + 8]), gA + (k) + 8);  \
    _Float16* sb = (tid < 128) ? &sBg[buf][0] : &sBu[buf][0];                \
    async_ld_b128(lds_off(sb + bRow * LDS_STRIDE + ldOff),     gB + (k));            \
    async_ld_b128(lds_off(sb + bRow * LDS_STRIDE + ldOff + 8), gB + (k) + 8);        \
  } while (0)

  G1_ISSUE(0, 0);
  int cur = 0;
  for (int k0 = 0; k0 < K; k0 += BK) {
    if (k0 + BK < K) {
      G1_ISSUE(cur ^ 1, k0 + BK);
      WAIT_ASYNC(4);                    // current buffer complete, next in flight
    } else {
      WAIT_ASYNC(0);
    }
    __syncthreads();

    v16h aF[2], gF[2], uF[2];
    for (int i = 0; i < 2; ++i)
      aF[i] = ld_frag(&sA[cur][(wm + i * 16 + fr) * LDS_STRIDE + kb]);
    for (int j = 0; j < 2; ++j) {
      gF[j] = ld_frag(&sBg[cur][(wn + j * 16 + fr) * LDS_STRIDE + kb]);
      uF[j] = ld_frag(&sBu[cur][(wn + j * 16 + fr) * LDS_STRIDE + kb]);
    }
    for (int i = 0; i < 2; ++i)
      for (int j = 0; j < 2; ++j) {
        accG[i][j] = __builtin_amdgcn_wmma_f32_16x16x32_f16(
            false, aF[i], false, gF[j], (short)0, accG[i][j], false, false);
        accU[i][j] = __builtin_amdgcn_wmma_f32_16x16x32_f16(
            false, aF[i], false, uF[j], (short)0, accU[i][j], false, false);
      }
    __syncthreads();                    // buffer free before next overwrite
    cur ^= 1;
  }
#undef G1_ISSUE

  // epilogue: h = silu(gate) * up  (C/D layout: M=r+(lane>=16?8:0), N=lane&15)
  int mloc = (lane >> 4) << 3;
  int nloc = lane & 15;
  for (int i = 0; i < 2; ++i)
    for (int j = 0; j < 2; ++j)
      for (int r = 0; r < 8; ++r) {
        float g = accG[i][j][r];
        float u = accU[i][j][r];
        float hv = g * (1.0f / (1.0f + expf(-g))) * u;
        size_t gm = (size_t)(m0 + wm + i * 16 + r + mloc);
        size_t gn = (size_t)(n0 + wn + j * 16 + nloc);
        H[gm * (size_t)Ndim + gn] = (_Float16)hv;
      }
}

// ---------------- GEMM 2: out = hq @ Wd^T, fp32 out ------------------------
// Per issue, every wave posts exactly 3 async b128 (2 for A, 1 for B tile).

__global__ __launch_bounds__(TPB)
void gemm2_single(const _Float16* __restrict__ A,   // [M,K] hq
                  const _Float16* __restrict__ B,   // [N,K] w_down fp16
                  float* __restrict__ Out,          // [M,N]
                  int M, int Ndim, int K) {
  __shared__ _Float16 sA[2][BM * LDS_STRIDE];
  __shared__ _Float16 sB[2][BN * LDS_STRIDE];

  int pm, pn;
  swizzle_pid(blockIdx.x, M / BM, Ndim / BN, pm, pn);
  int m0 = pm * BM, n0 = pn * BN;

  int tid = threadIdx.x, lane = tid & 31, wid = tid >> 5;
  int wm = (wid >> 1) * 32;
  int wn = (wid & 1) * 32;

  const v8f zero8 = {0.f,0.f,0.f,0.f,0.f,0.f,0.f,0.f};
  v8f acc[2][2];
  for (int i = 0; i < 2; ++i)
    for (int j = 0; j < 2; ++j) acc[i][j] = zero8;

  // A loader: row tid>>1, 32B; B loader: row tid>>2, 16B (all 256 threads)
  int aRow = tid >> 1;
  int aOff = (tid & 1) * 16;
  int bRow = tid >> 2;
  int bOff = (tid & 3) * 8;
  const _Float16* gA = A + (size_t)(m0 + aRow) * K + aOff;
  const _Float16* gB = B + (size_t)(n0 + bRow) * K + bOff;

  int fr = lane & 15;
  int kb = (lane >> 4) * 8;

#define G2_ISSUE(buf, k)                                                     \
  do {                                                                       \
    async_ld_b128(lds_off(&sA[buf][aRow * LDS_STRIDE + aOff]),     gA + (k));     \
    async_ld_b128(lds_off(&sA[buf][aRow * LDS_STRIDE + aOff + 8]), gA + (k) + 8); \
    async_ld_b128(lds_off(&sB[buf][bRow * LDS_STRIDE + bOff]),     gB + (k));     \
  } while (0)

  G2_ISSUE(0, 0);
  int cur = 0;
  for (int k0 = 0; k0 < K; k0 += BK) {
    if (k0 + BK < K) {
      G2_ISSUE(cur ^ 1, k0 + BK);
      WAIT_ASYNC(3);
    } else {
      WAIT_ASYNC(0);
    }
    __syncthreads();

    v16h aF[2], bF[2];
    for (int i = 0; i < 2; ++i)
      aF[i] = ld_frag(&sA[cur][(wm + i * 16 + fr) * LDS_STRIDE + kb]);
    for (int j = 0; j < 2; ++j)
      bF[j] = ld_frag(&sB[cur][(wn + j * 16 + fr) * LDS_STRIDE + kb]);
    for (int i = 0; i < 2; ++i)
      for (int j = 0; j < 2; ++j)
        acc[i][j] = __builtin_amdgcn_wmma_f32_16x16x32_f16(
            false, aF[i], false, bF[j], (short)0, acc[i][j], false, false);
    __syncthreads();
    cur ^= 1;
  }
#undef G2_ISSUE

  int mloc = (lane >> 4) << 3;
  int nloc = lane & 15;
  for (int i = 0; i < 2; ++i)
    for (int j = 0; j < 2; ++j)
      for (int r = 0; r < 8; ++r) {
        size_t gm = (size_t)(m0 + wm + i * 16 + r + mloc);
        size_t gn = (size_t)(n0 + wn + j * 16 + nloc);
        Out[gm * (size_t)Ndim + gn] = acc[i][j][r];
      }
}

// ---------------- launch -----------------

extern "C" void kernel_launch(void* const* d_in, const int* in_sizes, int n_in,
                              void* d_out, int out_size, void* d_ws, size_t ws_size,
                              hipStream_t stream) {
  const float* x       = (const float*)d_in[0];
  const float* w_gate  = (const float*)d_in[1];
  const float* w_up    = (const float*)d_in[2];
  const float* w_down  = (const float*)d_in[3];
  const int*   perm_up = (const int*)d_in[4];
  const int*   perm_dn = (const int*)d_in[5];
  const int*   sel_up  = (const int*)d_in[6];
  const int*   sel_dn  = (const int*)d_in[7];

  const int D = in_sizes[4];                    // 4096
  const int F = in_sizes[5];                    // 12288
  const long N = (long)in_sizes[0] / D;         // 8192 tokens

  // workspace carve-out (fp16 weights, fp16 activations, masks)
  char* ws = (char*)d_ws;
  const size_t szW = (size_t)F * D * sizeof(_Float16);
  _Float16* wg_h = (_Float16*)(ws);
  _Float16* wu_h = (_Float16*)(ws + szW);
  _Float16* wd_h = (_Float16*)(ws + 2 * szW);
  _Float16* xq   = (_Float16*)(ws + 3 * szW);
  _Float16* hbuf = (_Float16*)(ws + 3 * szW + (size_t)N * D * sizeof(_Float16));
  unsigned char* mu = (unsigned char*)(ws + 3 * szW
                       + (size_t)N * D * sizeof(_Float16)
                       + (size_t)N * F * sizeof(_Float16));
  unsigned char* md = mu + D;

  hipMemsetAsync(mu, 0, (size_t)(D + F), stream);

  f32_to_f16_kernel<<<8192, TPB, 0, stream>>>(w_gate, wg_h, (size_t)F * D);
  f32_to_f16_kernel<<<8192, TPB, 0, stream>>>(w_up,   wu_h, (size_t)F * D);
  f32_to_f16_kernel<<<8192, TPB, 0, stream>>>(w_down, wd_h, (size_t)D * F);
  build_masks_kernel<<<1, TPB, 0, stream>>>(perm_up, perm_dn, sel_up, sel_dn,
                                            mu, md, D, F);

  quant_x_kernel<<<(int)N, TPB, 0, stream>>>(x, mu, xq, D);

  int g1 = (int)(N / BM) * (F / BN);
  gemm1_dual_silu<<<g1, TPB, 0, stream>>>(xq, wg_h, wu_h, hbuf, (int)N, F, D);

  quant_h_kernel<<<(int)N, TPB, 0, stream>>>(hbuf, md, F);

  int g2 = (int)(N / BM) * (D / BN);
  gemm2_single<<<g2, TPB, 0, stream>>>(hbuf, wd_h, (float*)d_out, (int)N, D, F);
}
